// Tokenizer_50697793962315
// MI455X (gfx1250) — compile-verified, matrix operand order checked by reading, not run on hardware
//
#include <hip/hip_runtime.h>
#include <hip/hip_bf16.h>
#include <stdint.h>

// ---------------------------------------------------------------------------
// Hierarchical patch tokenizer for MI455X (gfx1250, wave32).
//  Stage A  : per-patch band sums + exact lower-median via 3-pass radix select
//             (LDS histograms, deterministic LDS reduction for band sums)
//  Stage B  : token_map build; per-batch bits table staged into LDS with
//             global_load_async_to_lds_b32 + s_wait_asynccnt  (CDNA5 async path)
//  Stage Q2 : per-batch 13-bin median
//  Stage B2 : in-place token_map -> |tm - q2| bytes
//  Stage C  : per-patch byte sums via v_wmma_i32_16x16x64_iu8 with ones-B
//             (1KB reduced per WMMA, A gathered with b64 loads)  (CDNA5 WMMA path)
//  Stage D  : stable descending rank of 143 scores -> ids_restore, token gather
// ---------------------------------------------------------------------------

#define BATCH 64
#define CH    3
#define IN    384
#define TOT   143   // 5 + 25 + 113

typedef int v8i __attribute__((ext_vector_type(8)));

// ---------------- Stage A: band sums + median, one block per (b,l,p) --------
template <int L>
__global__ __launch_bounds__(256) void stageA(const float* __restrict__ x,
                                              int* __restrict__ bits,
                                              int* __restrict__ tok)
{
    constexpr int PAc   = (L == 0) ? 2 : (L == 1) ? 4 : 8;
    constexpr int PSc   = IN / PAc;
    constexpr int BSc   = PSc / 6;
    constexpr int SHc   = PSc / 2;
    constexpr int P0c   = PAc * PAc;
    constexpr int FIRSTc = (L == 0) ? 0 : (L == 1) ? 5 : 30;
    constexpr int N     = CH * PSc * PSc;
    constexpr int RANK  = (N - 1) / 2;

    __shared__ int      hist[2048];
    __shared__ float    part[256 * 12];
    __shared__ float    bsum[12];
    __shared__ unsigned sel[2];
    __shared__ float    smed;

    const int tid = threadIdx.x;
    const int p   = blockIdx.x;
    const int b   = blockIdx.y;

    int y0, x0;
    if (p < P0c) { y0 = (p / PAc) * PSc;          x0 = (p % PAc) * PSc; }
    else { int q = p - P0c; y0 = SHc + (q / (PAc - 1)) * PSc; x0 = SHc + (q % (PAc - 1)) * PSc; }

    const float* xb = x + (size_t)b * CH * IN * IN;

    for (int i = tid; i < 2048; i += 256) hist[i] = 0;
    __syncthreads();

    // ---- pass 1: band sums + histogram of key bits [31:21] ----
    float av[6] = {0, 0, 0, 0, 0, 0}, ah[6] = {0, 0, 0, 0, 0, 0};
    for (int i = tid; i < N; i += 256) {
        int c   = i / (PSc * PSc);
        int rem = i - c * (PSc * PSc);
        int r   = rem / PSc;
        int col = rem - r * PSc;
        float v = xb[((size_t)c * IN + (y0 + r)) * IN + (x0 + col)];
        unsigned ub = __float_as_uint(v);
        unsigned u  = (ub & 0x80000000u) ? ~ub : (ub | 0x80000000u);
        atomicAdd(&hist[u >> 21], 1);
        int vb = r / BSc, hb = col / BSc;
#pragma unroll
        for (int k = 0; k < 6; k++) {
            av[k] += (vb == k) ? v : 0.0f;
            ah[k] += (hb == k) ? v : 0.0f;
        }
    }
#pragma unroll
    for (int k = 0; k < 6; k++) { part[tid * 12 + k] = av[k]; part[tid * 12 + 6 + k] = ah[k]; }
    __syncthreads();

    if (tid >= 16 && tid < 28) {            // deterministic band reduce
        int k = tid - 16; float s = 0.0f;
        for (int t = 0; t < 256; t++) s += part[t * 12 + k];
        bsum[k] = s;
    }
    if (tid == 0) {                         // select digit 1
        int cum = 0; unsigned bin = 0;
        for (; bin < 2048; bin++) { int h = hist[bin]; if (cum + h > RANK) break; cum += h; }
        sel[0] = bin; sel[1] = (unsigned)(RANK - cum);
    }
    __syncthreads();
    const unsigned b1 = sel[0];
    const int      r1 = (int)sel[1];
    __syncthreads();

    for (int i = tid; i < 2048; i += 256) hist[i] = 0;
    __syncthreads();

    // ---- pass 2: bits [20:10] among prefix matches ----
    for (int i = tid; i < N; i += 256) {
        int c   = i / (PSc * PSc);
        int rem = i - c * (PSc * PSc);
        int r   = rem / PSc;
        int col = rem - r * PSc;
        unsigned ub = __float_as_uint(xb[((size_t)c * IN + (y0 + r)) * IN + (x0 + col)]);
        unsigned u  = (ub & 0x80000000u) ? ~ub : (ub | 0x80000000u);
        if ((u >> 21) == b1) atomicAdd(&hist[(u >> 10) & 2047], 1);
    }
    __syncthreads();
    if (tid == 0) {
        int cum = 0; unsigned bin = 0;
        for (; bin < 2048; bin++) { int h = hist[bin]; if (cum + h > r1) break; cum += h; }
        sel[0] = bin; sel[1] = (unsigned)(r1 - cum);
    }
    __syncthreads();
    const unsigned b2 = sel[0];
    const int      r2 = (int)sel[1];
    __syncthreads();

    for (int i = tid; i < 1024; i += 256) hist[i] = 0;
    __syncthreads();

    // ---- pass 3: bits [9:0] ----
    const unsigned pref22 = (b1 << 11) | b2;
    for (int i = tid; i < N; i += 256) {
        int c   = i / (PSc * PSc);
        int rem = i - c * (PSc * PSc);
        int r   = rem / PSc;
        int col = rem - r * PSc;
        unsigned ub = __float_as_uint(xb[((size_t)c * IN + (y0 + r)) * IN + (x0 + col)]);
        unsigned u  = (ub & 0x80000000u) ? ~ub : (ub | 0x80000000u);
        if ((u >> 10) == pref22) atomicAdd(&hist[u & 1023], 1);
    }
    __syncthreads();
    if (tid == 0) {
        int cum = 0; unsigned bin = 0;
        for (; bin < 1024; bin++) { int h = hist[bin]; if (cum + h > r2) break; cum += h; }
        unsigned key = (b1 << 21) | (b2 << 10) | bin;
        unsigned fb  = (key & 0x80000000u) ? (key ^ 0x80000000u) : ~key;
        smed = __uint_as_float(fb);
    }
    __syncthreads();

    constexpr float INVCNT = 1.0f / (float)(CH * BSc * PSc);
    if (tid < 12) {
        float mean = bsum[tid] * INVCNT;
        bits[(size_t)(b * TOT + FIRSTc + p) * 12 + tid] = (mean < smed) ? 1 : 0;
    }
    if (tid == 0) {
        int t = 0;
#pragma unroll
        for (int k = 0; k < 12; k++)
            t += ((bsum[k] * INVCNT < smed) ? 1 : 0) << (11 - k);
        tok[b * TOT + FIRSTc + p] = t;
    }
}

// ---------------- Stage B: token_map + 13-bin histogram ---------------------
template <int PAc>
__device__ __forceinline__ int lvlContrib(const int* lb, int y, int x, int FIRSTc)
{
    constexpr int PSc = IN / PAc, BSc = PSc / 6, SHc = PSc / 2, P0c = PAc * PAc;
    int pid = FIRSTc + (y / PSc) * PAc + (x / PSc);
    int t   = lb[pid * 12 + (y % PSc) / BSc] + lb[pid * 12 + 6 + (x % PSc) / BSc];
    if (y >= SHc && y < IN - SHc && x >= SHc && x < IN - SHc) {
        int yy = y - SHc, xx = x - SHc;
        int pid2 = FIRSTc + P0c + (yy / PSc) * (PAc - 1) + (xx / PSc);
        t += lb[pid2 * 12 + (yy % PSc) / BSc] + lb[pid2 * 12 + 6 + (xx % PSc) / BSc];
    }
    return t;
}

__global__ __launch_bounds__(256) void stageB(const int* __restrict__ bits,
                                              unsigned char* __restrict__ tmap,
                                              int* __restrict__ hist13)
{
    __shared__ int lb[TOT * 12];
    __shared__ int lh[13];
    const int tid = threadIdx.x;
    const int b   = blockIdx.y;

    // CDNA5 async global->LDS staging of this batch's bits table
    const int* gb = bits + (size_t)b * TOT * 12;
    for (int i = tid; i < TOT * 12; i += 256) {
        unsigned   lo = (unsigned)(size_t)(&lb[i]);   // low 32 bits of flat = LDS offset
        const int* gp = gb + i;
        asm volatile("global_load_async_to_lds_b32 %0, %1, off"
                     :: "v"(lo), "v"(gp) : "memory");
    }
    if (tid < 13) lh[tid] = 0;
    asm volatile("s_wait_asynccnt 0" ::: "memory");
    __syncthreads();

    const int pix0 = blockIdx.x * 1024 + tid * 4;
    unsigned packed = 0;
#pragma unroll
    for (int j = 0; j < 4; j++) {
        int pix = pix0 + j;
        int y = pix / IN, xc = pix % IN;
        int tm = lvlContrib<2>(lb, y, xc, 0)
               + lvlContrib<4>(lb, y, xc, 5)
               + lvlContrib<8>(lb, y, xc, 30);
        atomicAdd(&lh[tm], 1);
        packed |= ((unsigned)tm) << (8 * j);
    }
    *(unsigned*)(tmap + (size_t)b * IN * IN + pix0) = packed;
    __syncthreads();
    if (tid < 13) atomicAdd(&hist13[b * 13 + tid], lh[tid]);
}

// ---------------- Stage Q2: per-batch lower median of token_map -------------
__global__ void stageQ2(const int* __restrict__ hist13, int* __restrict__ q2)
{
    int b = threadIdx.x;
    if (b >= BATCH) return;
    const int rr = (IN * IN - 1) / 2;   // 73727
    int cum = 0, v = 0;
    for (; v < 13; v++) { int h = hist13[b * 13 + v]; if (cum + h > rr) break; cum += h; }
    q2[b] = v;
}

// ---------------- Stage B2: in-place |tm - q2| per byte ---------------------
__global__ __launch_bounds__(256) void stageB2(unsigned char* __restrict__ tmap,
                                               const int* __restrict__ q2)
{
    const size_t i  = (size_t)blockIdx.x * 256 + threadIdx.x;   // dword index
    const size_t ib = i * 4;
    const int    b  = (int)(ib / (IN * IN));
    const int    q  = q2[b];
    unsigned t = *(const unsigned*)(tmap + ib);
    unsigned d = 0;
#pragma unroll
    for (int j = 0; j < 4; j++) {
        int tv = (t >> (8 * j)) & 0xFF;
        int dv = (tv > q) ? (tv - q) : (q - tv);
        d |= ((unsigned)dv) << (8 * j);
    }
    *(unsigned*)(tmap + ib) = d;
}

// ---------------- Stage C: per-patch byte sums via IU8 WMMA -----------------
// A-fragment gathered with b64 loads (all patch offsets are 8-byte aligned):
// each lane supplies 4 qwords = 32 bytes; one WMMA reduces 1024 bytes.
template <int L>
__global__ __launch_bounds__(256) void stageC(const unsigned char* __restrict__ dmap,
                                              int* __restrict__ score)
{
    constexpr int PAc = (L == 0) ? 2 : (L == 1) ? 4 : 8;
    constexpr int PSc = IN / PAc, SHc = PSc / 2, P0c = PAc * PAc;
    constexpr int PLc = P0c + (PAc - 1) * (PAc - 1);
    constexpr int FIRSTc = (L == 0) ? 0 : (L == 1) ? 5 : 30;
    constexpr int RQD = PSc / 8;        // qwords per patch row
    constexpr int NQ  = PSc * RQD;      // qwords per patch

    const int lane = threadIdx.x & 31;
    const int w    = blockIdx.x * 8 + (threadIdx.x >> 5);
    const int b    = w / PLc;
    const int p    = w - b * PLc;
    if (b >= BATCH) return;             // uniform (grids are exact)

    int y0, x0;
    if (p < P0c) { y0 = (p / PAc) * PSc;          x0 = (p % PAc) * PSc; }
    else { int q = p - P0c; y0 = SHc + (q / (PAc - 1)) * PSc; x0 = SHc + (q % (PAc - 1)) * PSc; }
    const unsigned char* dp0 = dmap + (size_t)b * IN * IN + (size_t)y0 * IN + x0;

    v8i ones;
#pragma unroll
    for (int r = 0; r < 8; r++) ones[r] = 0x01010101;
    v8i acc = {};

    for (int base = 0; base < NQ; base += 128) {
        v8i a;
#pragma unroll
        for (int r = 0; r < 4; r++) {
            int idx = base + r * 32 + lane;           // qword index within patch
            uint2 val = make_uint2(0u, 0u);
            if (idx < NQ) {
                int row = idx / RQD;
                int cq  = idx - row * RQD;
                unsigned off = (unsigned)row * IN + (unsigned)cq * 8;  // 32-bit offset
                val = *(const uint2*)(dp0 + off);
            }
            a[2 * r]     = (int)val.x;
            a[2 * r + 1] = (int)val.y;
        }
        // D = A(16x64 u8) * ones(64x16 u8) + C : 16 row sums; total = sum of D col 0
        acc = __builtin_amdgcn_wmma_i32_16x16x64_iu8(false, a, false, ones, acc, false, false);
    }
    int s = 0;
#pragma unroll
    for (int r = 0; r < 8; r++) s += acc[r];
    int tot = __shfl(s, 0, 32) + __shfl(s, 16, 32);   // M=0..7 (lane0) + M=8..15 (lane16)
    if (lane == 0) score[b * TOT + FIRSTc + p] = tot;
}

// ---------------- Stage D: stable descending ranks + token gather -----------
__global__ __launch_bounds__(192) void stageD(const int* __restrict__ score,
                                              const int* __restrict__ tok,
                                              int* __restrict__ out)
{
    __shared__ int ls[TOT];
    const int b = blockIdx.x, tid = threadIdx.x;
    if (tid < TOT) ls[tid] = score[b * TOT + tid];
    __syncthreads();
    if (tid < TOT) {
        int si = ls[tid], rank = 0;
        for (int j = 0; j < TOT; j++) {
            int sj = ls[j];
            rank += ((sj > si) || (sj == si && j < tid)) ? 1 : 0;
        }
        out[BATCH * 49 + b * TOT + tid] = rank;       // ids_restore
    }
    if (tid < 49) {
        int keep = (tid < 21) ? tid : tid + 9;        // 0-4, 5-20, 30-57
        out[b * 49 + tid] = tok[b * TOT + keep];      // token_keep
    }
}

__global__ void zeroK(int* __restrict__ p, int n)
{
    int i = blockIdx.x * blockDim.x + threadIdx.x;
    if (i < n) p[i] = 0;
}

// ---------------------------------------------------------------------------
extern "C" void kernel_launch(void* const* d_in, const int* in_sizes, int n_in,
                              void* d_out, int out_size, void* d_ws, size_t ws_size,
                              hipStream_t stream)
{
    const float* x = (const float*)d_in[0];
    char* ws = (char*)d_ws;

    int*           bits   = (int*)(ws + 0);                 // 64*143*12*4 = 439296
    int*           tok    = (int*)(ws + 439296);            // 64*143*4    = 36608
    unsigned char* tmap   = (unsigned char*)(ws + 475904);  // 64*384*384  = 9437184
    int*           hist13 = (int*)(ws + 9913088);           // 64*13*4     = 3328
    int*           q2     = (int*)(ws + 9916416);           // 64*4
    int*           score  = (int*)(ws + 9916672);           // 64*143*4    = 36608
    int*           out    = (int*)d_out;                    // 3136 + 9152 ints

    zeroK<<<4, 256, 0, stream>>>(hist13, BATCH * 13);

    stageA<0><<<dim3(5, BATCH),   256, 0, stream>>>(x, bits, tok);
    stageA<1><<<dim3(25, BATCH),  256, 0, stream>>>(x, bits, tok);
    stageA<2><<<dim3(113, BATCH), 256, 0, stream>>>(x, bits, tok);

    stageB<<<dim3(IN * IN / 1024, BATCH), 256, 0, stream>>>(bits, tmap, hist13);
    stageQ2<<<1, 64, 0, stream>>>(hist13, q2);
    stageB2<<<(BATCH * IN * IN / 4) / 256, 256, 0, stream>>>(tmap, q2);

    stageC<0><<<(BATCH * 5)   / 8, 256, 0, stream>>>(tmap, score);
    stageC<1><<<(BATCH * 25)  / 8, 256, 0, stream>>>(tmap, score);
    stageC<2><<<(BATCH * 113) / 8, 256, 0, stream>>>(tmap, score);

    stageD<<<BATCH, 192, 0, stream>>>(score, tok, out);
}
